// SharedEncoder_87909390615182
// MI455X (gfx1250) — matrix operand order repaired and checked by reading, not compiled, not run on hardware
//
#include <hip/hip_runtime.h>

typedef __attribute__((ext_vector_type(16))) _Float16 v16h;
typedef __attribute__((ext_vector_type(8)))  float    v8f;

// Device-scope hardware float atomic add (global_atomic_add_f32, no CAS loop).
__device__ __forceinline__ void atomAddF(float* p, float v) {
  __hip_atomic_fetch_add(p, v, __ATOMIC_RELAXED, __HIP_MEMORY_SCOPE_AGENT);
}

// ---------------------------------------------------------------------------
// Graph preprocessing (once per call): degrees, rsqrt, per-edge norms
// ---------------------------------------------------------------------------
__global__ void k_deg_init(float* deg, int n) {
  int i = blockIdx.x * blockDim.x + threadIdx.x;
  if (i < n) deg[i] = 1.0f;   // self-loop contributes 1 to in-degree
}

// edge_index is int64 [2][E]; values < 2^31 so read the low 32-bit word.
__global__ void k_deg_count(const int* __restrict__ ei, float* deg, int E) {
  int e = blockIdx.x * blockDim.x + threadIdx.x;
  if (e < E) {
    int d = ei[2 * (E + e)];           // low word of edge_index[1][e]
    atomAddF(&deg[d], 1.0f);
  }
}

__global__ void k_make_dis(float* degdis, int n) {
  int i = blockIdx.x * blockDim.x + threadIdx.x;
  if (i < n) degdis[i] = rsqrtf(degdis[i]);   // deg >= 1 always
}

__global__ void k_edge_norm(const int* __restrict__ ei,
                            const float* __restrict__ dis,
                            float* __restrict__ norm, int E) {
  int e = blockIdx.x * blockDim.x + threadIdx.x;
  if (e < E) {
    int s = ei[2 * e];
    int d = ei[2 * (E + e)];
    norm[e] = dis[s] * dis[d];
  }
}

// ---------------------------------------------------------------------------
// Node-feature linear transform via V_WMMA_F32_16X16X32_F16.
// One wave owns a 16-row node tile; K padded to 32, N chunked by 16.
// A layout (16-bit 16x32): lanes 0-15 row M=l, VGPR v<4 -> K=8*half+2v,
// v>=4 -> K=16+8*half+2(v-4).  B (32x16): same K per lane/VGPR as A,
// column = lane&15.  C/D: VGPR v -> row 8*half+v, col = lane&15.
// ---------------------------------------------------------------------------
template <int FIN, int FOUT>
__global__ void k_linear_wmma(const float* __restrict__ in,
                              const float* __restrict__ W,
                              float* __restrict__ out, int nrows) {
  const int wave  = threadIdx.x >> 5;
  const int lane  = threadIdx.x & 31;
  const int tile  = blockIdx.x * (blockDim.x >> 5) + wave;
  const int ntile = (nrows + 15) >> 4;
  if (tile >= ntile) return;                 // wave-uniform exit

  const int half = lane >> 4;
  const int l    = lane & 15;
  int row = tile * 16 + l;
  if (row > nrows - 1) row = nrows - 1;      // clamp (loads only)
  const float* rp = in + (size_t)row * FIN;

  v16h a;
#pragma unroll
  for (int v = 0; v < 8; ++v) {
    const int k0 = (v < 4) ? (8 * half + 2 * v) : (16 + 8 * half + 2 * (v - 4));
    a[2 * v]     = (k0     < FIN) ? (_Float16)rp[k0]     : (_Float16)0.0f;
    a[2 * v + 1] = (k0 + 1 < FIN) ? (_Float16)rp[k0 + 1] : (_Float16)0.0f;
  }

  const bool fulltile = ((tile + 1) * 16 <= nrows);   // wave-uniform

#pragma unroll
  for (int nb = 0; nb < FOUT; nb += 16) {
    const int col = nb + l;
    v16h b;
#pragma unroll
    for (int v = 0; v < 8; ++v) {
      const int k0 = (v < 4) ? (8 * half + 2 * v) : (16 + 8 * half + 2 * (v - 4));
      b[2 * v]     = (k0     < FIN && col < FOUT) ? (_Float16)W[k0 * FOUT + col]       : (_Float16)0.0f;
      b[2 * v + 1] = (k0 + 1 < FIN && col < FOUT) ? (_Float16)W[(k0 + 1) * FOUT + col] : (_Float16)0.0f;
    }
    v8f c = {};
    v8f d = __builtin_amdgcn_wmma_f32_16x16x32_f16(
        /*neg_a=*/false, a, /*neg_b=*/false, b,
        /*c_mod=*/(short)0, c, /*reuse_a=*/false, /*reuse_b=*/false);

    // One 64-bit base, compile-time immediate offsets for the 8 row stores.
    float* op = out + ((size_t)tile * 16 + 8 * half) * FOUT + col;
    if (fulltile) {
      if (FOUT >= 16 || col < FOUT) {
#pragma unroll
        for (int v = 0; v < 8; ++v) op[v * FOUT] = d[v];
      }
    } else {
#pragma unroll
      for (int v = 0; v < 8; ++v) {
        const int r = tile * 16 + 8 * half + v;
        if ((FOUT >= 16 || col < FOUT) && r < nrows) op[v * FOUT] = d[v];
      }
    }
  }
}

// ---------------------------------------------------------------------------
// Aggregation: self-loop init, edge scatter-add, bias+ReLU
// ---------------------------------------------------------------------------
template <int F>
__global__ void k_self_init(const float* __restrict__ h,
                            const float* __restrict__ dis,
                            float* __restrict__ out, int n) {
  const int CP = F / 4;
  int idx = blockIdx.x * blockDim.x + threadIdx.x;
  if (idx >= n * CP) return;
  int i = idx / CP;
  float s = dis[i];
  s = s * s;                                 // self-loop norm = dis[i]^2
  float4 hv = ((const float4*)h)[idx];
  float4 r;
  r.x = hv.x * s; r.y = hv.y * s; r.z = hv.z * s; r.w = hv.w * s;
  ((float4*)out)[idx] = r;
}

template <int F>
__global__ void k_edge_scatter(const int* __restrict__ ei,
                               const float* __restrict__ norm,
                               const float* __restrict__ h,
                               float* out, int E) {
  const int CPE = F / 4;                     // float4 chunks per edge
  int idx = blockIdx.x * blockDim.x + threadIdx.x;
  if (idx >= E * CPE) return;
  int e = idx / CPE;
  int c = idx % CPE;
  int s = ei[2 * e];
  int d = ei[2 * (E + e)];
  float w = norm[e];
  float4 hv = *(const float4*)(h + (size_t)s * F + 4 * c);
  float* o = out + (size_t)d * F + 4 * c;
  atomAddF(o + 0, hv.x * w);
  atomAddF(o + 1, hv.y * w);
  atomAddF(o + 2, hv.z * w);
  atomAddF(o + 3, hv.w * w);
}

template <int F>
__global__ void k_bias_relu(float* out, const float* __restrict__ b, int n) {
  int idx = blockIdx.x * blockDim.x + threadIdx.x;
  if (idx >= n * F) return;
  float v = out[idx] + b[idx & (F - 1)];
  out[idx] = v > 0.0f ? v : 0.0f;
}

// ---------------------------------------------------------------------------
// Mean pool over 64 graphs: LDS (ds_add_f32) block bins, then global atomics
// ---------------------------------------------------------------------------
__global__ void k_zero(float* p, int n) {
  int i = blockIdx.x * blockDim.x + threadIdx.x;
  if (i < n) p[i] = 0.0f;
}

__global__ void k_pool(const int* __restrict__ batch, const float* __restrict__ h,
                       float* sums, float* cnts, int n) {
  __shared__ float ls[64 * 8];
  __shared__ float lc[64];
  for (int j = threadIdx.x; j < 64 * 8; j += blockDim.x) ls[j] = 0.0f;
  for (int j = threadIdx.x; j < 64; j += blockDim.x) lc[j] = 0.0f;
  __syncthreads();
  int i = blockIdx.x * blockDim.x + threadIdx.x;
  if (i < n) {
    int g = batch[2 * i];                    // low word of int64 batch id
#pragma unroll
    for (int f = 0; f < 8; ++f) atomicAdd(&ls[g * 8 + f], h[(size_t)i * 8 + f]);
    atomicAdd(&lc[g], 1.0f);
  }
  __syncthreads();
  for (int j = threadIdx.x; j < 64 * 8; j += blockDim.x)
    if (ls[j] != 0.0f) atomAddF(&sums[j], ls[j]);
  for (int j = threadIdx.x; j < 64; j += blockDim.x)
    if (lc[j] != 0.0f) atomAddF(&cnts[j], lc[j]);
}

__global__ void k_final(const float* __restrict__ sums, const float* __restrict__ cnts,
                        const float* __restrict__ Wf, const float* __restrict__ bf,
                        float* out) {
  int t = threadIdx.x;
  if (t >= 64 * 3) return;
  int g = t / 3, o = t % 3;
  float c = cnts[g];
  if (c < 1.0f) c = 1.0f;
  float acc = bf[o];
#pragma unroll
  for (int k = 0; k < 8; ++k) acc += (sums[g * 8 + k] / c) * Wf[k * 3 + o];
  out[t] = acc;
}

// ---------------------------------------------------------------------------
// Launch
// ---------------------------------------------------------------------------
static inline int divup(int a, int b) { return (a + b - 1) / b; }

extern "C" void kernel_launch(void* const* d_in, const int* in_sizes, int n_in,
                              void* d_out, int out_size, void* d_ws, size_t ws_size,
                              hipStream_t stream) {
  const float* x    = (const float*)d_in[0];
  const int*   ei   = (const int*)d_in[1];   // int64 data, read low words
  const int*   bat  = (const int*)d_in[2];   // int64 data, read low words
  const float* W1   = (const float*)d_in[3];
  const float* b1   = (const float*)d_in[4];
  const float* W2   = (const float*)d_in[5];
  const float* b2   = (const float*)d_in[6];
  const float* W3   = (const float*)d_in[7];
  const float* b3   = (const float*)d_in[8];
  const float* Wf   = (const float*)d_in[9];
  const float* bf   = (const float*)d_in[10];
  float*       out  = (float*)d_out;

  const int N = in_sizes[0] / 3;             // 100000
  const int E = in_sizes[1] / 2;             // 3200000

  float* ws   = (float*)d_ws;
  float* dis  = ws;                                        // N floats (deg -> dis)
  float* nrm  = dis  + (((size_t)N + 255) & ~(size_t)255); // E floats
  float* bufH = nrm  + (((size_t)E + 255) & ~(size_t)255); // N*32 floats
  float* bufA = bufH + (size_t)N * 32;                     // N*32 floats
  float* sums = bufA + (size_t)N * 32;                     // 64*8 floats
  float* cnts = sums + 64 * 8;                             // 64 floats

  const int B = 256;

  // Degree / normalization (graph-only, once per call)
  k_deg_init <<<divup(N, B), B, 0, stream>>>(dis, N);
  k_deg_count<<<divup(E, B), B, 0, stream>>>(ei, dis, E);
  k_make_dis <<<divup(N, B), B, 0, stream>>>(dis, N);
  k_edge_norm<<<divup(E, B), B, 0, stream>>>(ei, dis, nrm, E);

  const int gridW = divup(divup(N, 16), B / 32);           // 1 wave per 16-row tile

  // Layer 1: x(3) -> 32
  k_linear_wmma<3, 32><<<gridW, B, 0, stream>>>(x, W1, bufH, N);
  k_self_init<32>    <<<divup(N * 8, B), B, 0, stream>>>(bufH, dis, bufA, N);
  k_edge_scatter<32> <<<divup(E, B / 8), B, 0, stream>>>(ei, nrm, bufH, bufA, E);
  k_bias_relu<32>    <<<divup(N * 32, B), B, 0, stream>>>(bufA, b1, N);

  // Layer 2: 32 -> 16
  k_linear_wmma<32, 16><<<gridW, B, 0, stream>>>(bufA, W2, bufH, N);
  k_self_init<16>    <<<divup(N * 4, B), B, 0, stream>>>(bufH, dis, bufA, N);
  k_edge_scatter<16> <<<divup(E, B / 4), B, 0, stream>>>(ei, nrm, bufH, bufA, E);
  k_bias_relu<16>    <<<divup(N * 16, B), B, 0, stream>>>(bufA, b2, N);

  // Layer 3: 16 -> 8
  k_linear_wmma<16, 8><<<gridW, B, 0, stream>>>(bufA, W3, bufH, N);
  k_self_init<8>     <<<divup(N * 2, B), B, 0, stream>>>(bufH, dis, bufA, N);
  k_edge_scatter<8>  <<<divup(E, B / 2), B, 0, stream>>>(ei, nrm, bufH, bufA, E);
  k_bias_relu<8>     <<<divup(N * 8, B), B, 0, stream>>>(bufA, b3, N);

  // Mean pool + final linear
  k_zero <<<divup(64 * 9, B), B, 0, stream>>>(sums, 64 * 9);   // sums + cnts
  k_pool <<<divup(N, B), B, 0, stream>>>(bat, bufA, sums, cnts, N);
  k_final<<<1, 192, 0, stream>>>(sums, cnts, Wf, bf, out);
}
